// CTBlock_21930103014257
// MI455X (gfx1250) — compile-verified, matrix operand order checked by reading, not run on hardware
//
#include <hip/hip_runtime.h>
#include <hip/hip_bf16.h>
#include <math.h>

// ---------------------------------------------------------------------------
// CDNA5 / gfx1250 implementation. wave32, WMMA f32_16x16x32_f16 everywhere.
// Layout convention: activations [rows][K] row-major, weights [N][K] row-major,
// so both A and B WMMA fragments are two contiguous 16-byte loads per lane.
// ---------------------------------------------------------------------------

typedef _Float16 v8h  __attribute__((ext_vector_type(8)));
typedef _Float16 v16h __attribute__((ext_vector_type(16)));
typedef float    v8f  __attribute__((ext_vector_type(8)));

#define BB 8
#define SS 1024
#define CC 512
#define HH 8
#define DD 64
#define FF 2048
#define LL 3
#define BS (BB*SS)                 // 8192
#define BSC ((size_t)BS*CC)        // 4194304

// Fragment loader: matrix stored [rows][ld] f16, returns the 16x32 fragment
// with row0..row0+15 and k0..k0+31 per the CDNA5 16-bit A/B VGPR layout:
//   lane l: row = row0 + (l&15);  element j -> k = (j&7) + ((j>>3)<<4) + ((l>>4)<<3)
// i.e. halves j=0..7 at k0+kb.., j=8..15 at k0+kb+16..  (two b128 loads).
__device__ __forceinline__ v16h frag_ld(const _Float16* base, int ld, int row0, int k0) {
    int lane = threadIdx.x & 31;
    int r  = row0 + (lane & 15);
    int kb = k0 + ((lane >> 4) << 3);
    const _Float16* p = base + (size_t)r * ld + kb;
    v8h lo = *(const v8h*)p;
    v8h hi = *(const v8h*)(p + 16);
    v16h f;
#pragma unroll
    for (int j = 0; j < 8; ++j) { f[j] = lo[j]; f[j + 8] = hi[j]; }
    return f;
}

__device__ __forceinline__ v8f wmma32(v16h a, v16h b, v8f c) {
    return __builtin_amdgcn_wmma_f32_16x16x32_f16(false, a, false, b, (short)0, c, false, false);
}

__device__ __forceinline__ void zero8(v8f& v) {
#pragma unroll
    for (int i = 0; i < 8; ++i) v[i] = 0.0f;
}

// ---------------------------------------------------------------------------
// small elementwise kernels
// ---------------------------------------------------------------------------
__global__ void k_copy_f32(float* __restrict__ dst, const float* __restrict__ src) {
    size_t i = (size_t)blockIdx.x * 256 + threadIdx.x;
    dst[i] = src[i];
}

__global__ void k_cvt_f16(_Float16* __restrict__ dst, const float* __restrict__ src, int n) {
    int i = blockIdx.x * 256 + threadIdx.x;
    if (i < n) dst[i] = (_Float16)src[i];
}

// conv_w (L,C,C,3) fp32 -> wp [L][t][co][ci] f16
__global__ void k_pack_convw(_Float16* __restrict__ wp, const float* __restrict__ cw) {
    int i = blockIdx.x * 256 + threadIdx.x;          // total L*3*C*C = 2359296
    int l   = i / (3 * CC * CC);
    int rem = i % (3 * CC * CC);
    int t   = rem / (CC * CC);
    int co  = (rem >> 9) & (CC - 1);
    int ci  = rem & (CC - 1);
    wp[i] = (_Float16)cw[(((size_t)(l * CC + co)) * CC + ci) * 3 + t];
}

// h fp32 [B,S,C] -> hp f16 [B,S+2,C] with zero rows at s=0 and s=S+1
__global__ void k_pad_f16(_Float16* __restrict__ hp, const float* __restrict__ h) {
    size_t i = (size_t)blockIdx.x * 256 + threadIdx.x;   // total B*(S+2)*C = 4202496
    int c = (int)(i & (CC - 1));
    size_t row = i >> 9;
    int sr = (int)(row % (SS + 2));
    int b  = (int)(row / (SS + 2));
    float v = (sr == 0 || sr == SS + 1) ? 0.0f
              : h[((size_t)b * SS + (sr - 1)) * CC + c];
    hp[i] = (_Float16)v;
}

// GroupNorm stats: one block per (b, group); 64 channels x 1024 positions
__global__ void k_gn_stats(const float* __restrict__ y, float* __restrict__ mean,
                           float* __restrict__ rstd) {
    int bg = blockIdx.x;              // 0..63
    int b = bg >> 3, g = bg & 7;
    const float* base = y + (size_t)b * SS * CC + g * 64;
    float s = 0.f, s2 = 0.f;
    for (int t = threadIdx.x; t < 64 * SS; t += 256) {
        int sr = t >> 6, cc2 = t & 63;
        float v = base[(size_t)sr * CC + cc2];
        s += v; s2 += v * v;
    }
    __shared__ float rs[256], rq[256];
    rs[threadIdx.x] = s; rq[threadIdx.x] = s2;
    __syncthreads();
    for (int st = 128; st > 0; st >>= 1) {
        if (threadIdx.x < st) { rs[threadIdx.x] += rs[threadIdx.x + st];
                                rq[threadIdx.x] += rq[threadIdx.x + st]; }
        __syncthreads();
    }
    if (threadIdx.x == 0) {
        float m = rs[0] / (64.0f * SS);
        float v = rq[0] / (64.0f * SS) - m * m;
        mean[bg] = m;
        rstd[bg] = rsqrtf(v + 1e-5f);
    }
}

// h += gelu(groupnorm(y))
__global__ void k_gn_gelu_res(float* __restrict__ h, const float* __restrict__ y,
                              const float* __restrict__ mean, const float* __restrict__ rstd,
                              const float* __restrict__ gg, const float* __restrict__ gb) {
    size_t i = (size_t)blockIdx.x * 256 + threadIdx.x;   // BSC total
    int c = (int)(i & (CC - 1));
    int b = (int)(i >> 19);                               // / (S*C) = / 2^19
    int bg = b * 8 + (c >> 6);
    float xn = (y[i] - mean[bg]) * rstd[bg] * gg[c] + gb[c];
    float ge = 0.5f * xn * (1.0f + erff(xn * 0.70710678118654752f));
    h[i] += ge;
}

// h *= 1/3 (in place), x0h = f16(h)
__global__ void k_scale_cvt(float* __restrict__ h, _Float16* __restrict__ x0h) {
    size_t i = (size_t)blockIdx.x * 256 + threadIdx.x;
    float v = h[i] * (1.0f / 3.0f);
    h[i] = v;
    x0h[i] = (_Float16)v;
}

// qkv fp32 [B,S,3C] -> qf/kf [B,H,S,D] f16, vT [B,H,D,S] f16
__global__ void k_split_qkv(const float* __restrict__ qkv, _Float16* __restrict__ qf,
                            _Float16* __restrict__ kf, _Float16* __restrict__ vT) {
    size_t i = (size_t)blockIdx.x * 256 + threadIdx.x;   // BS*3C total
    int col = (int)(i % (3 * CC));
    size_t row = i / (3 * CC);
    int b = (int)(row >> 10), s = (int)(row & (SS - 1));
    _Float16 hv = (_Float16)qkv[i];
    int which = col >> 9;
    int hh = (col >> 6) & 7;
    int d  = col & 63;
    size_t bh = (size_t)(b * HH + hh);
    if (which == 0)      qf[(bh * SS + s) * DD + d] = hv;
    else if (which == 1) kf[(bh * SS + s) * DD + d] = hv;
    else                 vT[(bh * DD + d) * SS + s] = hv;
}

// ---------------------------------------------------------------------------
// Generic WMMA GEMM: out[M][N] = A[M][K] @ B[N][K]^T + bias, opt ReLU, f16 out
// block = 256 thr (8 waves); block tile 32x256, each wave a 16x64 tile.
// grid = (N/256, M/32)
// ---------------------------------------------------------------------------
template <bool RELU, bool F16OUT>
__global__ __launch_bounds__(256) void k_gemm(const _Float16* __restrict__ A,
                                              const _Float16* __restrict__ Bm,
                                              const float* __restrict__ bias,
                                              float* __restrict__ outF,
                                              _Float16* __restrict__ outH,
                                              int N, int K) {
    int wv = threadIdx.x >> 5, lane = threadIdx.x & 31;
    int m0 = blockIdx.y * 32 + (wv >> 2) * 16;
    int n0 = blockIdx.x * 256 + (wv & 3) * 64;
    v8f acc[4];
#pragma unroll
    for (int j = 0; j < 4; ++j) zero8(acc[j]);

    for (int k0 = 0; k0 < K; k0 += 32) {
        v16h a = frag_ld(A, K, m0, k0);
#pragma unroll
        for (int j = 0; j < 4; ++j) {
            v16h b = frag_ld(Bm, K, n0 + 16 * j, k0);
            acc[j] = wmma32(a, b, acc[j]);
        }
    }
    int mo = (lane >> 4) << 3, nc = lane & 15;
#pragma unroll
    for (int j = 0; j < 4; ++j)
#pragma unroll
        for (int r = 0; r < 8; ++r) {
            int m = m0 + mo + r;
            int n = n0 + 16 * j + nc;
            float v = acc[j][r] + bias[n];
            if (RELU) v = fmaxf(v, 0.0f);
            size_t oi = (size_t)m * N + n;
            if (F16OUT) outH[oi] = (_Float16)v; else outF[oi] = v;
        }
}

// ---------------------------------------------------------------------------
// Conv1d(k=3) as 3 accumulated GEMM passes over padded activations hp.
// y[b,s,co] = sum_t sum_ci wp[t][co][ci] * hp[b][s+t][ci] + bias[co]
// grid = (512/256=2, 8192/32=256)
// ---------------------------------------------------------------------------
__global__ __launch_bounds__(256) void k_conv_gemm(float* __restrict__ y,
                                                   const _Float16* __restrict__ hp,
                                                   const _Float16* __restrict__ wp_l,
                                                   const float* __restrict__ bias) {
    int wv = threadIdx.x >> 5, lane = threadIdx.x & 31;
    int m0 = blockIdx.y * 32 + (wv >> 2) * 16;       // row in [0, BS)
    int n0 = blockIdx.x * 256 + (wv & 3) * 64;
    int b = m0 >> 10, s = m0 & (SS - 1);
    v8f acc[4];
#pragma unroll
    for (int j = 0; j < 4; ++j) zero8(acc[j]);

    for (int t = 0; t < 3; ++t) {
        const _Float16* Arow = hp + ((size_t)(b * (SS + 2) + s + t)) * CC;
        const _Float16* Bw = wp_l + (size_t)t * CC * CC;
        for (int k0 = 0; k0 < CC; k0 += 32) {
            v16h a = frag_ld(Arow, CC, 0, k0);
#pragma unroll
            for (int j = 0; j < 4; ++j) {
                v16h bf = frag_ld(Bw, CC, n0 + 16 * j, k0);
                acc[j] = wmma32(a, bf, acc[j]);
            }
        }
    }
    int mo = (lane >> 4) << 3, nc = lane & 15;
#pragma unroll
    for (int j = 0; j < 4; ++j)
#pragma unroll
        for (int r = 0; r < 8; ++r) {
            int m = m0 + mo + r;
            int n = n0 + 16 * j + nc;
            y[(size_t)m * CC + n] = acc[j][r] + bias[n];
        }
}

// ---------------------------------------------------------------------------
// Attention: one block per (b*H+h, 16-query tile). 8 waves.
// scores (QK^T/8, mask) -> LDS fp32 -> softmax -> attn out + f16 probs in LDS
// -> ctx = P @ V via K-split across waves -> LDS reduce -> ctxh [B,S,H,D]
// dynamic LDS: 64KB scores + 32KB probs + reduce scratch
// ---------------------------------------------------------------------------
__global__ __launch_bounds__(256) void k_attn(const _Float16* __restrict__ qf,
                                              const _Float16* __restrict__ kf,
                                              const _Float16* __restrict__ vT,
                                              const unsigned char* __restrict__ pm,
                                              float* __restrict__ attn_out,
                                              _Float16* __restrict__ ctxh) {
    extern __shared__ char smem[];
    float*    sc   = (float*)smem;                                  // 16*1024 f32
    _Float16* pr   = (_Float16*)(smem + 16 * 1024 * 4);             // 16*1024 f16
    float*    red  = (float*)(smem + 16 * 1024 * 4 + 16 * 1024 * 2);// 256
    float*    rowv = red + 256;                                     // 16

    int bh = blockIdx.y;
    int q0 = blockIdx.x * 16;
    int b  = bh >> 3;
    int tid = threadIdx.x;
    int wv = tid >> 5, lane = tid & 31;
    int mo = (lane >> 4) << 3, nc = lane & 15;

    const _Float16* qb = qf + (size_t)bh * SS * DD;
    const _Float16* kb = kf + (size_t)bh * SS * DD;
    const _Float16* vb = vT + (size_t)bh * DD * SS;
    const float NEGINF = -__builtin_inff();

    // ---- scores: wave w covers key cols [w*128, w*128+128) ----
    for (int nt = 0; nt < 8; ++nt) {
        int n0 = wv * 128 + nt * 16;
        v8f acc; zero8(acc);
        v16h a = frag_ld(qb, DD, q0, 0);
        v16h bf = frag_ld(kb, DD, n0, 0);
        acc = wmma32(a, bf, acc);
        a  = frag_ld(qb, DD, q0, 32);
        bf = frag_ld(kb, DD, n0, 32);
        acc = wmma32(a, bf, acc);
#pragma unroll
        for (int r = 0; r < 8; ++r) {
            int m = mo + r, n = n0 + nc;
            float v = acc[r] * 0.125f;                       // 1/sqrt(64)
            if (pm[b * SS + n]) v = NEGINF;
            sc[m * SS + n] = v;
        }
    }
    __syncthreads();

    // ---- softmax over each of the 16 rows (16 threads / row) ----
    int row = tid >> 4, jj = tid & 15;
    float mx = NEGINF;
    for (int kk = jj; kk < SS; kk += 16) mx = fmaxf(mx, sc[row * SS + kk]);
    red[row * 16 + jj] = mx;
    __syncthreads();
    if (jj == 0) {
        float m2 = red[row * 16];
        for (int z = 1; z < 16; ++z) m2 = fmaxf(m2, red[row * 16 + z]);
        rowv[row] = m2;
    }
    __syncthreads();
    float rmax = rowv[row];
    float sum = 0.f;
    for (int kk = jj; kk < SS; kk += 16) {
        float e = __expf(sc[row * SS + kk] - rmax);
        sc[row * SS + kk] = e;
        sum += e;
    }
    __syncthreads();
    red[row * 16 + jj] = sum;
    __syncthreads();
    if (jj == 0) {
        float s3 = 0.f;
        for (int z = 0; z < 16; ++z) s3 += red[row * 16 + z];
        rowv[row] = s3;
    }
    __syncthreads();
    float inv = 1.0f / rowv[row];
    for (int kk = jj; kk < SS; kk += 16) {
        float p = sc[row * SS + kk] * inv;
        pr[row * SS + kk] = (_Float16)p;
        attn_out[((size_t)bh * SS + q0 + row) * SS + kk] = p;
    }
    __syncthreads();

    // ---- ctx = P @ V, K split across 8 waves (128 keys each) ----
    v8f a2[4];
#pragma unroll
    for (int j = 0; j < 4; ++j) zero8(a2[j]);
    int kbase = wv * 128;
    for (int ks = 0; ks < 128; ks += 32) {
        v16h a = frag_ld(pr, SS, 0, kbase + ks);
#pragma unroll
        for (int dt = 0; dt < 4; ++dt) {
            v16h bf = frag_ld(vb, SS, dt * 16, kbase + ks);
            a2[dt] = wmma32(a, bf, a2[dt]);
        }
    }
    float* cred = sc;   // reuse score region: 8 waves * 16 * 64 = 8192 floats
#pragma unroll
    for (int dt = 0; dt < 4; ++dt)
#pragma unroll
        for (int r = 0; r < 8; ++r) {
            int m = mo + r, d = dt * 16 + nc;
            cred[(wv * 16 + m) * 64 + d] = a2[dt][r];
        }
    __syncthreads();
    int hh = bh & 7;
    for (int i = tid; i < 16 * 64; i += 256) {
        int m = i >> 6, d = i & 63;
        float s4 = 0.f;
        for (int w = 0; w < 8; ++w) s4 += cred[(w * 16 + m) * 64 + d];
        ctxh[(((size_t)(b * SS + q0 + m)) * HH + hh) * DD + d] = (_Float16)s4;
    }
}

// ---------------------------------------------------------------------------
// GEMM (N=512) + bias + residual + LayerNorm fused.
// block = 256 thr, tile 16 x 512 (8 waves x 64 cols). grid = M/16 = 512.
// ---------------------------------------------------------------------------
__global__ __launch_bounds__(256) void k_gemm_ln(float* __restrict__ outF,
                                                 _Float16* __restrict__ outH,
                                                 const _Float16* __restrict__ A,
                                                 const _Float16* __restrict__ Bm,
                                                 const float* __restrict__ bias,
                                                 const float* __restrict__ res,
                                                 const float* __restrict__ gamma,
                                                 const float* __restrict__ beta,
                                                 int K) {
    __shared__ float tile[16 * CC];
    __shared__ float redS[256], redQ[256];
    __shared__ float rowm[16], rowr[16];

    int tid = threadIdx.x, wv = tid >> 5, lane = tid & 31;
    int m0 = blockIdx.x * 16;
    int n0 = wv * 64;
    v8f acc[4];
#pragma unroll
    for (int j = 0; j < 4; ++j) zero8(acc[j]);

    for (int k0 = 0; k0 < K; k0 += 32) {
        v16h a = frag_ld(A, K, m0, k0);
#pragma unroll
        for (int j = 0; j < 4; ++j) {
            v16h b = frag_ld(Bm, K, n0 + 16 * j, k0);
            acc[j] = wmma32(a, b, acc[j]);
        }
    }
    int mo = (lane >> 4) << 3, nc = lane & 15;
#pragma unroll
    for (int j = 0; j < 4; ++j)
#pragma unroll
        for (int r = 0; r < 8; ++r) {
            int m = mo + r;
            int n = n0 + 16 * j + nc;
            tile[m * CC + n] = acc[j][r] + bias[n] + res[((size_t)(m0 + m)) * CC + n];
        }
    __syncthreads();

    int row = tid >> 4, jj = tid & 15;
    float s = 0.f, s2 = 0.f;
    for (int kk = jj; kk < CC; kk += 16) {
        float v = tile[row * CC + kk];
        s += v; s2 += v * v;
    }
    redS[row * 16 + jj] = s; redQ[row * 16 + jj] = s2;
    __syncthreads();
    if (jj == 0) {
        float S = 0.f, S2 = 0.f;
        for (int z = 0; z < 16; ++z) { S += redS[row * 16 + z]; S2 += redQ[row * 16 + z]; }
        float mean = S / (float)CC;
        float var = S2 / (float)CC - mean * mean;
        rowm[row] = mean;
        rowr[row] = rsqrtf(var + 1e-5f);
    }
    __syncthreads();
    for (int i = tid; i < 16 * CC; i += 256) {
        int m = i >> 9, n = i & (CC - 1);
        float v = (tile[i] - rowm[m]) * rowr[m] * gamma[n] + beta[n];
        size_t oi = ((size_t)(m0 + m)) * CC + n;
        outF[oi] = v;
        if (outH) outH[oi] = (_Float16)v;
    }
}

// ---------------------------------------------------------------------------
// host side
// ---------------------------------------------------------------------------
extern "C" void kernel_launch(void* const* d_in, const int* in_sizes, int n_in,
                              void* d_out, int out_size, void* d_ws, size_t ws_size,
                              hipStream_t stream) {
    const float* x_in   = (const float*)d_in[0];
    const unsigned char* pm = (const unsigned char*)d_in[1];
    const float* conv_w = (const float*)d_in[2];
    const float* conv_b = (const float*)d_in[3];
    const float* gn_g   = (const float*)d_in[4];
    const float* gn_b   = (const float*)d_in[5];
    const float* inpw   = (const float*)d_in[6];
    const float* inpb   = (const float*)d_in[7];
    const float* outw   = (const float*)d_in[8];
    const float* outb   = (const float*)d_in[9];
    const float* f1w    = (const float*)d_in[10];
    const float* f1b    = (const float*)d_in[11];
    const float* f2w    = (const float*)d_in[12];
    const float* f2b    = (const float*)d_in[13];
    const float* ln1g   = (const float*)d_in[14];
    const float* ln1b   = (const float*)d_in[15];
    const float* ln2g   = (const float*)d_in[16];
    const float* ln2b   = (const float*)d_in[17];

    float* out_x    = (float*)d_out;              // [B,S,C]
    float* out_attn = out_x + BSC;                // [B,H,S,S]

    // workspace bump allocator (256B aligned)
    char* ws = (char*)d_ws;
    size_t off = 0;
    auto alloc = [&](size_t bytes) -> void* {
        void* p = ws + off;
        off += (bytes + 255) & ~(size_t)255;
        return p;
    };
    const size_t HP = (size_t)BB * (SS + 2) * CC;     // 4202496
    const size_t WP = (size_t)LL * 3 * CC * CC;       // 2359296

    float*    h    = (float*)alloc(BSC * 4);          // activations / x0 (in place)
    float*    y    = (float*)alloc(BSC * 4);          // conv out; reused as x1
    _Float16* hp   = (_Float16*)alloc(HP * 2);        // padded f16; reused as ctxh
    _Float16* wp   = (_Float16*)alloc(WP * 2);
    float*    mean = (float*)alloc(64 * 4);
    float*    rstd = (float*)alloc(64 * 4);
    _Float16* x0h  = (_Float16*)alloc(BSC * 2);
    _Float16* wq   = (_Float16*)alloc((size_t)3 * CC * CC * 2);
    _Float16* wo   = (_Float16*)alloc((size_t)CC * CC * 2);
    _Float16* w1   = (_Float16*)alloc((size_t)FF * CC * 2);
    _Float16* w2   = (_Float16*)alloc((size_t)CC * FF * 2);
    float*    qkv  = (float*)alloc((size_t)BS * 3 * CC * 4);
    _Float16* qf   = (_Float16*)alloc(BSC * 2);
    _Float16* kf   = (_Float16*)alloc(BSC * 2);
    _Float16* vT   = (_Float16*)alloc(BSC * 2);
    _Float16* x1h  = (_Float16*)alloc(BSC * 2);
    _Float16* tb   = (_Float16*)alloc((size_t)BS * FF * 2);
    _Float16* ctxh = hp;     // reuse (hp dead after conv loop)
    float*    x1   = y;      // reuse (y dead after conv loop)

    // ---- weight conversion (once per launch) ----
    k_pack_convw<<<(int)(WP / 256), 256, 0, stream>>>(wp, conv_w);
    k_cvt_f16<<<(3 * CC * CC + 255) / 256, 256, 0, stream>>>(wq, inpw, 3 * CC * CC);
    k_cvt_f16<<<(CC * CC + 255) / 256, 256, 0, stream>>>(wo, outw, CC * CC);
    k_cvt_f16<<<(FF * CC + 255) / 256, 256, 0, stream>>>(w1, f1w, FF * CC);
    k_cvt_f16<<<(CC * FF + 255) / 256, 256, 0, stream>>>(w2, f2w, CC * FF);

    // ---- CNN block ----
    k_copy_f32<<<(int)(BSC / 256), 256, 0, stream>>>(h, x_in);
    for (int l = 0; l < LL; ++l) {
        k_pad_f16<<<(int)(HP / 256), 256, 0, stream>>>(hp, h);
        k_conv_gemm<<<dim3(CC / 256, BS / 32), 256, 0, stream>>>(
            y, hp, wp + (size_t)l * 3 * CC * CC, conv_b + l * CC);
        k_gn_stats<<<64, 256, 0, stream>>>(y, mean, rstd);
        k_gn_gelu_res<<<(int)(BSC / 256), 256, 0, stream>>>(
            h, y, mean, rstd, gn_g + l * CC, gn_b + l * CC);
    }
    k_scale_cvt<<<(int)(BSC / 256), 256, 0, stream>>>(h, x0h);   // h := x0

    // ---- QKV projection: [8192,512] @ [1536,512]^T ----
    k_gemm<false, false><<<dim3(3 * CC / 256, BS / 32), 256, 0, stream>>>(
        x0h, wq, inpb, qkv, nullptr, 3 * CC, CC);
    k_split_qkv<<<(int)((size_t)BS * 3 * CC / 256), 256, 0, stream>>>(qkv, qf, kf, vT);

    // ---- attention (writes attn to d_out and ctxh) ----
    size_t attn_lds = (size_t)16 * 1024 * 4 + 16 * 1024 * 2 + (256 + 16) * 4;
    k_attn<<<dim3(SS / 16, BB * HH), 256, attn_lds, stream>>>(
        qf, kf, vT, pm, out_attn, ctxh);

    // ---- out-proj + residual(x0) + LN1 -> x1 (f32 + f16) ----
    k_gemm_ln<<<BS / 16, 256, 0, stream>>>(
        x1, x1h, ctxh, wo, outb, h, ln1g, ln1b, CC);

    // ---- FFN1 with ReLU -> f16 ----
    k_gemm<true, true><<<dim3(FF / 256, BS / 32), 256, 0, stream>>>(
        x1h, w1, f1b, nullptr, tb, FF, CC);

    // ---- FFN2 + residual(x1) + LN2 -> d_out x ----
    k_gemm_ln<<<BS / 16, 256, 0, stream>>>(
        out_x, (_Float16*)nullptr, tb, w2, f2b, x1, ln2g, ln2b, FF);

    (void)in_sizes; (void)n_in; (void)out_size; (void)ws_size;
}
